// MoEBase_68255620268371
// MI455X (gfx1250) — matrix operand order repaired and checked by reading, not compiled, not run on hardware
//
#include <hip/hip_runtime.h>

typedef __attribute__((ext_vector_type(16))) unsigned short v16u;
typedef __attribute__((ext_vector_type(16))) __bf16         v16bf;
typedef __attribute__((ext_vector_type(8)))  float          v8f;
typedef __attribute__((ext_vector_type(4)))  unsigned int   v4u;
typedef __attribute__((ext_vector_type(8)))  int            v8i;
typedef __attribute__((ext_vector_type(4)))  int            v4i;

#define D_DIM 1024
#define H_DIM 512
#define E_NUM 8

// LDS strides (elements) chosen for bank-conflict-free b128 fragment loads
#define SX_STR 1032   // 16 x-rows,  m*516 dwords -> 4m mod 64, all distinct
#define SH_STR 520    // 16 h-rows,  m*260 dwords -> 4m mod 64, all distinct
#define SWS    40     // weight chunk [N][K=32] rows (64B data + 16B pad), n*20 mod 64 distinct

__device__ __forceinline__ unsigned short f2bf(float f) {
  unsigned int u = __float_as_uint(f);
  u += 0x7FFFu + ((u >> 16) & 1u);      // round to nearest even
  return (unsigned short)(u >> 16);
}

__device__ __forceinline__ v8f vzero() {
  v8f z;
#pragma unroll
  for (int i = 0; i < 8; ++i) z[i] = 0.0f;
  return z;
}

__device__ __forceinline__ v8f wmma_bf16(const v16u& a, const v16u& b, v8f c) {
  return __builtin_amdgcn_wmma_f32_16x16x32_bf16(
      false, __builtin_bit_cast(v16bf, a),
      false, __builtin_bit_cast(v16bf, b),
      (short)0, c, false, false);
}

// A-fragment: 16 consecutive K at fixed M (K = i + g*16, matching B)
__device__ __forceinline__ v16u ld_frag_a(const unsigned short* p) {
  struct U { uint4 a, b; } u;
  u.a = *(const uint4*)p;
  u.b = *(const uint4*)(p + 8);
  return __builtin_bit_cast(v16u, u);
}
// B-fragment: 16 consecutive K at fixed N (contiguous K order, same as A, K = i + g*16)
__device__ __forceinline__ v16u ld_frag_b(const unsigned short* p) {
  struct U { uint4 a, b; } u;
  u.a = *(const uint4*)p;
  u.b = *(const uint4*)(p + 8);
  return __builtin_bit_cast(v16u, u);
}

// wait for TENSORcnt <= 1 (older of two outstanding TDM ops retired) or == 0
__device__ __forceinline__ void tdm_wait(bool keep_one) {
  if (keep_one) __builtin_amdgcn_s_wait_tensorcnt(1);
  else          __builtin_amdgcn_s_wait_tensorcnt(0);
}

// ------------------------------------------------------------------ TDM load
// 2D tile DMA: global (bf16, dim0-contiguous) -> LDS, 16B pad after each 64B
// row so LDS rows land on the SWS=40-element padded stride.
__device__ __forceinline__ void tdm_load_2d(const unsigned short* gsrc, void* lds,
                                            unsigned tensor_d0, unsigned tensor_d1,
                                            unsigned tile_d0, unsigned tile_d1,
                                            unsigned stride_d0) {
  unsigned long long ga = (unsigned long long)gsrc;
  unsigned int la = (unsigned int)(unsigned long long)lds;   // LDS byte offset
  v4u g0;
  g0[0] = 1u;                                  // count=1 (valid user descriptor)
  g0[1] = la;                                  // lds_addr
  g0[2] = (unsigned int)ga;                    // global_addr[31:0]
  g0[3] = (unsigned int)(ga >> 32) | (2u << 30);  // global_addr[56:32] | type=2
  v8i g1;
  g1[0] = (1 << 16)                            // data_size = 2 bytes
        | (1 << 20)                            // pad_enable
        | (3 << 22)                            // pad_interval: 16 DWORDs (64B)
        | (3 << 25);                           // pad_amount:   4 DWORDs (16B)
  g1[1] = (int)(tensor_d0 << 16);              // tensor_dim0[15:0]
  g1[2] = (int)((tensor_d0 >> 16) | (tensor_d1 << 16));
  g1[3] = (int)((tensor_d1 >> 16) | (tile_d0 << 16));
  g1[4] = (int)tile_d1;                        // tile_dim1 (tile_dim2 = 0)
  g1[5] = (int)stride_d0;                      // tensor_dim0_stride[31:0]
  g1[6] = 0;                                   // stride0 hi, stride1 lo
  g1[7] = 0;                                   // stride1 hi
  v4i g2 = {0, 0, 0, 0};
  v4i g3 = {0, 0, 0, 0};
#if __clang_major__ >= 23
  v8i gz = {0, 0, 0, 0, 0, 0, 0, 0};
  __builtin_amdgcn_tensor_load_to_lds(g0, g1, g2, g3, gz, 0);
#else
  __builtin_amdgcn_tensor_load_to_lds(g0, g1, g2, g3, 0);
#endif
}

// ---------------------------------------------------------------- zero / init
__global__ void moe_zero(float* __restrict__ out, size_t n, int* __restrict__ counts) {
  size_t i = (size_t)blockIdx.x * blockDim.x + threadIdx.x;
  size_t stride = (size_t)gridDim.x * blockDim.x;
  for (; i < n; i += stride) out[i] = 0.0f;
  if (blockIdx.x == 0 && threadIdx.x < E_NUM) counts[threadIdx.x] = 0;
}

// ------------------------------------------------- f32 -> bf16 with transpose
// src: [E][K][N] f32 (N contiguous)  ->  dst: [E][N][K] bf16 (K contiguous)
__global__ __launch_bounds__(256) void moe_cvt_t(const float* __restrict__ src,
                                                 unsigned short* __restrict__ dst,
                                                 int K, int N) {
  __shared__ unsigned short tile[32][33];
  const int e   = blockIdx.z;
  const int k0  = blockIdx.x * 32;
  const int n0  = blockIdx.y * 32;
  const int tid = threadIdx.x;
  {
    const int row = tid >> 3;           // 0..31 (K)
    const int c4  = tid & 7;            // 0..7  (N/4)
    float4 v = *(const float4*)(src + ((size_t)(e * K + k0 + row) * N + n0 + c4 * 4));
    tile[row][c4 * 4 + 0] = f2bf(v.x);
    tile[row][c4 * 4 + 1] = f2bf(v.y);
    tile[row][c4 * 4 + 2] = f2bf(v.z);
    tile[row][c4 * 4 + 3] = f2bf(v.w);
  }
  __syncthreads();
  {
    const int nrow = tid >> 3;          // 0..31 (N)
    const int k4   = tid & 7;           // 0..7  (K/4)
    unsigned int lo = (unsigned int)tile[k4 * 4 + 0][nrow] |
                      ((unsigned int)tile[k4 * 4 + 1][nrow] << 16);
    unsigned int hi = (unsigned int)tile[k4 * 4 + 2][nrow] |
                      ((unsigned int)tile[k4 * 4 + 3][nrow] << 16);
    uint2 o; o.x = lo; o.y = hi;
    *(uint2*)(dst + (size_t)(e * N + n0 + nrow) * K + k0 + k4 * 4) = o;
  }
}

// ---------------------------------------------------------------- router
__global__ __launch_bounds__(256) void moe_router(
    const float* __restrict__ x, const float* __restrict__ wr,
    unsigned short* __restrict__ xb,
    int* __restrict__ counts, int* __restrict__ tids, float* __restrict__ gates,
    int T) {
  const int lane = threadIdx.x & 31;
  const int wid  = threadIdx.x >> 5;
  const int t = blockIdx.x * 8 + wid;
  if (t >= T) return;

  const float* xr = x + (size_t)t * D_DIM;
  unsigned short* xbr = xb + (size_t)t * D_DIM;

  float acc[E_NUM];
#pragma unroll
  for (int e = 0; e < E_NUM; ++e) acc[e] = 0.0f;

  for (int d = lane; d < D_DIM; d += 32) {
    float xv = xr[d];
    xbr[d] = f2bf(xv);
#pragma unroll
    for (int e = 0; e < E_NUM; ++e) acc[e] += xv * wr[d * E_NUM + e];
  }
#pragma unroll
  for (int e = 0; e < E_NUM; ++e) {
#pragma unroll
    for (int off = 16; off > 0; off >>= 1)
      acc[e] += __shfl_xor(acc[e], off, 32);
  }

  if (lane == 0) {
    float mx = acc[0];
#pragma unroll
    for (int e = 1; e < E_NUM; ++e) mx = fmaxf(mx, acc[e]);
    float p[E_NUM], s = 0.0f;
#pragma unroll
    for (int e = 0; e < E_NUM; ++e) { p[e] = __expf(acc[e] - mx); s += p[e]; }
    const float inv = 1.0f / s;
    int i1 = 0; float v1 = p[0];
#pragma unroll
    for (int e = 1; e < E_NUM; ++e) if (p[e] > v1) { v1 = p[e]; i1 = e; }
    int i2 = -1; float v2 = -1.0f;
#pragma unroll
    for (int e = 0; e < E_NUM; ++e) if (e != i1 && p[e] > v2) { v2 = p[e]; i2 = e; }

    int pos1 = atomicAdd(&counts[i1], 1);
    tids[i1 * T + pos1]  = t;
    gates[i1 * T + pos1] = v1 * inv;
    int pos2 = atomicAdd(&counts[i2], 1);
    tids[i2 * T + pos2]  = t;
    gates[i2 * T + pos2] = v2 * inv;
  }
}

// ---------------------------------------------------------------- expert FFN
// Weights pre-transposed: w1b/w3b = [E][H][D], w2b = [E][D][H] (K fastest).
// One block = 16 expert-assignments of expert blockIdx.y; 8 waves.
// TDM double-buffering: the two 40KB halves of sw ping-pong so each DMA
// overlaps WMMA compute on the other half (s_wait_tensorcnt 1 / 0).
__global__ __launch_bounds__(256) void moe_ffn(
    const unsigned short* __restrict__ xb,
    const unsigned short* __restrict__ w1b,
    const unsigned short* __restrict__ w3b,
    const unsigned short* __restrict__ w2b,
    const int* __restrict__ counts,
    const int* __restrict__ tids,
    const float* __restrict__ gates,
    float* __restrict__ out, int T) {
  const int e    = blockIdx.y;
  const int cnt  = counts[e];
  const int base = blockIdx.x * 16;
  if (base >= cnt) return;

  __shared__ __align__(16) unsigned short sx[16 * SX_STR];   // ~32 KB token tile
  __shared__ __align__(16) unsigned short sw[1024 * SWS];    // 2 x 40 KB chunk buffers
  __shared__ __align__(16) unsigned short sh[16 * SH_STR];   // ~16 KB hidden tile
  __shared__ int   s_tid[16];
  __shared__ float s_gw[16];
  __shared__ int   s_tok[16];

  unsigned short* const sw0 = sw;
  unsigned short* const sw1 = sw + 512 * SWS;

  const int tid  = threadIdx.x;
  const int lane = tid & 31;
  const int wid  = tid >> 5;
  const int m    = lane & 15;   // row/col within 16-wide tile
  const int g    = lane >> 4;   // half-wave group

  if (tid < 16) {
    int r  = base + tid;
    int tk = (r < cnt) ? tids[e * T + r] : -1;
    s_tid[tid] = tk;
    s_gw[tid]  = (r < cnt) ? gates[e * T + r] : 0.0f;
    s_tok[tid] = (tk >= 0) ? tk : 0;
  }
  __syncthreads();

  const unsigned short* w1e = w1b + (size_t)e * H_DIM * D_DIM;
  const unsigned short* w3e = w3b + (size_t)e * H_DIM * D_DIM;
  const unsigned short* w2e = w2b + (size_t)e * D_DIM * H_DIM;

  // kick off first w1/w3 chunk DMAs; they overlap the x-row gather below
  if (wid == 0) {
    tdm_load_2d(w1e, sw0, D_DIM, H_DIM, 32, H_DIM, D_DIM);
    tdm_load_2d(w3e, sw1, D_DIM, H_DIM, 32, H_DIM, D_DIM);
  }

  // gather 16 bf16 token rows into LDS (uint4 = 8 bf16), padded row stride
  {
    const uint4* src = (const uint4*)xb;
    uint4* dst = (uint4*)sx;
#pragma unroll
    for (int it = 0; it < 8; ++it) {
      int li  = tid + it * 256;        // 0..2047
      int row = li >> 7;               // 128 data uint4 per row
      int c8  = li & 127;
      dst[row * (SX_STR / 8) + c8] = src[(size_t)s_tok[row] * (D_DIM / 8) + c8];
    }
  }

  // ---- pass 1: A1 = x@w1 (sw0), A3 = x@w3 (sw1), ping-pong DMA/compute ----
  v8f acc1[4], acc3[4];
#pragma unroll
  for (int t = 0; t < 4; ++t) { acc1[t] = vzero(); acc3[t] = vzero(); }

  for (int kk = 0; kk < D_DIM; kk += 32) {
    const bool next = (kk + 32) < D_DIM;
    // --- half A: w1 chunk in sw0 ---
    if (wid == 0) tdm_wait(true);                         // w1(kk) landed
    __syncthreads();                                      // (also fences sx gather)
    v16u af = ld_frag_a(sx + m * SX_STR + kk + g * 16);
#pragma unroll
    for (int t = 0; t < 4; ++t) {
      const int n = wid * 64 + t * 16 + m;
      acc1[t] = wmma_bf16(af, ld_frag_b(sw0 + n * SWS + g * 16), acc1[t]);
    }
    __syncthreads();                                      // done reading sw0
    if (wid == 0 && next)
      tdm_load_2d(w1e + kk + 32, sw0, D_DIM, H_DIM, 32, H_DIM, D_DIM);
    // --- half B: w3 chunk in sw1 ---
    if (wid == 0) tdm_wait(next);                         // w3(kk) landed
    __syncthreads();
#pragma unroll
    for (int t = 0; t < 4; ++t) {
      const int n = wid * 64 + t * 16 + m;
      acc3[t] = wmma_bf16(af, ld_frag_b(sw1 + n * SWS + g * 16), acc3[t]);
    }
    __syncthreads();                                      // done reading sw1
    if (wid == 0 && next)
      tdm_load_2d(w3e + kk + 32, sw1, D_DIM, H_DIM, 32, H_DIM, D_DIM);
  }

  // kick off first w2 chunk DMAs (N-halves); they overlap the SwiGLU below
  if (wid == 0) {
    tdm_load_2d(w2e, sw0, H_DIM, D_DIM, 32, H_DIM, H_DIM);                      // N 0..511
    tdm_load_2d(w2e + (size_t)512 * H_DIM, sw1, H_DIM, D_DIM, 32, H_DIM, H_DIM); // N 512..1023
  }

  // ---- SwiGLU: h = silu(A1) * A3 -> LDS bf16 ----
#pragma unroll
  for (int t = 0; t < 4; ++t) {
#pragma unroll
    for (int r = 0; r < 8; ++r) {
      int row = r + g * 8;
      int col = wid * 64 + t * 16 + m;
      float a = acc1[t][r];
      float hv = (a / (1.0f + __expf(-a))) * acc3[t][r];
      sh[row * SH_STR + col] = f2bf(hv);
    }
  }

  // ---- pass 2: Y = h @ w2; sw0 = N-half [0,512), sw1 = N-half [512,1024) ----
  v8f accy[8];
#pragma unroll
  for (int t = 0; t < 8; ++t) accy[t] = vzero();

  for (int kk = 0; kk < H_DIM; kk += 32) {
    const bool next = (kk + 32) < H_DIM;
    // --- N-half 0 in sw0 ---
    if (wid == 0) tdm_wait(true);
    __syncthreads();                                      // (also fences sh writes)
    v16u af = ld_frag_a(sh + m * SH_STR + kk + g * 16);
#pragma unroll
    for (int t = 0; t < 4; ++t) {
      const int n = wid * 64 + t * 16 + m;                // N in [0,512)
      accy[t] = wmma_bf16(af, ld_frag_b(sw0 + n * SWS + g * 16), accy[t]);
    }
    __syncthreads();
    if (wid == 0 && next)
      tdm_load_2d(w2e + kk + 32, sw0, H_DIM, D_DIM, 32, H_DIM, H_DIM);
    // --- N-half 1 in sw1 ---
    if (wid == 0) tdm_wait(next);
    __syncthreads();
#pragma unroll
    for (int t = 0; t < 4; ++t) {
      const int n = wid * 64 + t * 16 + m;                // N-512 row in sw1
      accy[4 + t] = wmma_bf16(af, ld_frag_b(sw1 + n * SWS + g * 16), accy[4 + t]);
    }
    __syncthreads();
    if (wid == 0 && next)
      tdm_load_2d(w2e + (size_t)512 * H_DIM + kk + 32, sw1, H_DIM, D_DIM, 32, H_DIM, H_DIM);
  }

  // ---- gate-scale + accumulate into output ----
#pragma unroll
  for (int r = 0; r < 8; ++r) {
    int row = r + g * 8;
    int tk  = s_tid[row];
    if (tk < 0) continue;
    float gw = s_gw[row];
    float* orow = out + (size_t)tk * D_DIM;
#pragma unroll
    for (int t = 0; t < 8; ++t) {
      int col = (t < 4 ? 0 : 512) + wid * 64 + (t & 3) * 16 + m;
      atomicAdd(&orow[col], gw * accy[t][r]);
    }
  }
}

// ---------------------------------------------------------------- launcher
extern "C" void kernel_launch(void* const* d_in, const int* in_sizes, int n_in,
                              void* d_out, int out_size, void* d_ws, size_t ws_size,
                              hipStream_t stream) {
  (void)n_in; (void)out_size; (void)ws_size;
  const float* x  = (const float*)d_in[0];
  const float* wr = (const float*)d_in[1];
  const float* w1 = (const float*)d_in[2];
  const float* w2 = (const float*)d_in[3];
  const float* w3 = (const float*)d_in[4];
  // d_in[5] = top_k (assumed 2, matching the reference setup)

  const int T = in_sizes[0] / D_DIM;                 // B*S tokens
  const size_t nw = (size_t)E_NUM * D_DIM * H_DIM;   // elems per weight tensor

  char* ws = (char*)d_ws;
  unsigned short* xb  = (unsigned short*)ws;  ws += (size_t)T * D_DIM * 2;
  unsigned short* w1b = (unsigned short*)ws;  ws += nw * 2;
  unsigned short* w3b = (unsigned short*)ws;  ws += nw * 2;
  unsigned short* w2b = (unsigned short*)ws;  ws += nw * 2;
  int*   counts = (int*)ws;                   ws += 64;
  int*   tids   = (int*)ws;                   ws += (size_t)E_NUM * T * sizeof(int);
  float* gates  = (float*)ws;

  float* out = (float*)d_out;
  const size_t nout = (size_t)T * D_DIM;

  moe_zero<<<4096, 256, 0, stream>>>(out, nout, counts);
  // transpose-convert weights to K-contiguous bf16 layouts
  moe_cvt_t<<<dim3(D_DIM / 32, H_DIM / 32, E_NUM), 256, 0, stream>>>(w1, w1b, D_DIM, H_DIM);
  moe_cvt_t<<<dim3(D_DIM / 32, H_DIM / 32, E_NUM), 256, 0, stream>>>(w3, w3b, D_DIM, H_DIM);
  moe_cvt_t<<<dim3(H_DIM / 32, D_DIM / 32, E_NUM), 256, 0, stream>>>(w2, w2b, H_DIM, D_DIM);
  moe_router<<<(T + 7) / 8, 256, 0, stream>>>(x, wr, xb, counts, tids, gates, T);
  dim3 grid((T + 15) / 16, E_NUM);
  moe_ffn<<<grid, 256, 0, stream>>>(xb, w1b, w3b, w2b, counts, tids, gates, out, T);
}
